// SequenceQuantizerSoftEMA_89867895701686
// MI455X (gfx1250) — compile-verified, matrix operand order checked by reading, not run on hardware
//
#include <hip/hip_runtime.h>
#include <math.h>

// ---------------------------------------------------------------------------
// VQ soft-EMA quantizer, fused for MI455X (gfx1250, wave32, WMMA + TDM).
//   inputs          : f32 [64,512,512]  -> N=32768 rows, D=512
//   padding_mask    : bool[64,512]      (1 byte per element assumed)
//   codebook_weight : f32 [1024,512]
// outputs (concat in d_out):
//   quantized_st f32[64,512,512] | samples f32[64,512,1024] | loss | perplexity
// ---------------------------------------------------------------------------

typedef __attribute__((ext_vector_type(16))) _Float16     v16h;
typedef __attribute__((ext_vector_type(8)))  _Float16     v8h;
typedef __attribute__((ext_vector_type(8)))  float        v8f;
typedef __attribute__((ext_vector_type(4)))  float        v4f;
typedef __attribute__((ext_vector_type(4)))  unsigned int v4u;
typedef __attribute__((ext_vector_type(8)))  int          v8i;
typedef __attribute__((ext_vector_type(4)))  int          v4i;

#define K_CODES     1024
#define D_MODEL     512
#define N_ROWS      32768
#define NUM_SAMPLES 10

// workspace layout (as floats from base, except cb16 region)
#define WS_SSE        0
#define WS_NONPAD     1
#define WS_COUNTS     64            // 1024 floats
#define WS_C2         (64 + 1024)   // 1024 floats
#define WS_CB16_BYTES 16384         // _Float16[1024*512] starts here (1 MB)

// LDS layout (dynamic) for the main kernel
#define LDS_DIST_BYTES  (16 * K_CODES * 4)   // 64 KB f32 dist tile
#define LDS_STAGE_BYTES (16 * D_MODEL * 4)   // 32 KB f32 x tile (TDM target)
#define LDS_TOTAL_BYTES (LDS_DIST_BYTES + LDS_STAGE_BYTES + 64)

// ---------------------------------------------------------------------------
__device__ __forceinline__ float waveReduceSum(float v) {
  #pragma unroll
  for (int o = 16; o > 0; o >>= 1) v += __shfl_xor(v, o, 32);
  return v;
}
__device__ __forceinline__ float waveReduceMax(float v) {
  #pragma unroll
  for (int o = 16; o > 0; o >>= 1) v = fmaxf(v, __shfl_xor(v, o, 32));
  return v;
}

__device__ __forceinline__ unsigned rotl32(unsigned x, int n) {
  return (x << n) | (x >> (32 - n));
}

// Threefry-2x32, 20 rounds: counter-based deterministic PRNG.
__device__ __forceinline__ void threefry2x32(unsigned k0, unsigned k1,
                                             unsigned c0, unsigned c1,
                                             unsigned& o0, unsigned& o1) {
  const unsigned ks0 = k0, ks1 = k1, ks2 = k0 ^ k1 ^ 0x1BD11BDAu;
  unsigned x0 = c0 + ks0, x1 = c1 + ks1;
  const int ra[4] = {13, 15, 26, 6};
  const int rb[4] = {17, 29, 16, 24};
  #pragma unroll
  for (int j = 0; j < 4; j++) { x0 += x1; x1 = rotl32(x1, ra[j]); x1 ^= x0; }
  x0 += ks1; x1 += ks2 + 1u;
  #pragma unroll
  for (int j = 0; j < 4; j++) { x0 += x1; x1 = rotl32(x1, rb[j]); x1 ^= x0; }
  x0 += ks2; x1 += ks0 + 2u;
  #pragma unroll
  for (int j = 0; j < 4; j++) { x0 += x1; x1 = rotl32(x1, ra[j]); x1 ^= x0; }
  x0 += ks0; x1 += ks1 + 3u;
  #pragma unroll
  for (int j = 0; j < 4; j++) { x0 += x1; x1 = rotl32(x1, rb[j]); x1 ^= x0; }
  x0 += ks1; x1 += ks2 + 4u;
  #pragma unroll
  for (int j = 0; j < 4; j++) { x0 += x1; x1 = rotl32(x1, ra[j]); x1 ^= x0; }
  x0 += ks2; x1 += ks0 + 5u;
  o0 = x0; o1 = x1;
}

// ---------------------------------------------------------------------------
// Tensor Data Mover: DMA a [rows x 512] f32 tile (contiguous, stride 512)
// from global memory into LDS at byte offset ldsOff. One wave issues this;
// caller must s_wait_tensorcnt then barrier. D# bitfields per CDNA5 ISA §8.
__device__ __forceinline__ void tdm_load_f32_tile_to_lds(unsigned ldsOff,
                                                         const void* gptr,
                                                         unsigned rows) {
  const unsigned long long ga = (unsigned long long)(uintptr_t)gptr;
  v4u g0;
  g0[0] = 1u;                                    // count=1, user descriptor
  g0[1] = ldsOff;                                // lds_addr (bytes)
  g0[2] = (unsigned)(ga & 0xFFFFFFFFu);          // global_addr[31:0]
  g0[3] = (unsigned)((ga >> 32) & 0x01FFFFFFu)   // global_addr[56:32]
          | (2u << 30);                          // type = 2 ("image")
  v8i g1;
  g1[0] = (int)(2u << 16);        // workgroup_mask=0, data_size=2 (4 bytes)
  g1[1] = (int)(512u << 16);      // tensor_dim0[15:0] = 512  (bits 79:48)
  g1[2] = (int)(rows << 16);      // tensor_dim0 hi = 0, tensor_dim1 lo = rows
  g1[3] = (int)(512u << 16);      // tensor_dim1 hi = 0, tile_dim0 = 512
  g1[4] = (int)rows;              // tile_dim1 = rows, tile_dim2 = 0
  g1[5] = 512;                    // tensor_dim0_stride[31:0] = 512 elements
  g1[6] = 0;                      // stride hi / tensor_dim1_stride lo
  g1[7] = 0;                      // tensor_dim1_stride hi
  v4i gz; gz[0] = gz[1] = gz[2] = gz[3] = 0;     // 2D: groups 2/3 unused
#if __has_include(<hip/amd_detail/amd_gfx1250_TDM.h>)
  // amdgpu-toolchain (clang-23 / therock-10.0): 6-arg builtin
  v8i gz8; gz8[0]=gz8[1]=gz8[2]=gz8[3]=gz8[4]=gz8[5]=gz8[6]=gz8[7]=0;
  __builtin_amdgcn_tensor_load_to_lds(g0, g1, gz, gz, gz8, 0);
#else
  // ROCm 7.2 (clang-22): 5-arg builtin
  __builtin_amdgcn_tensor_load_to_lds(g0, g1, gz, gz, 0);
#endif
}

// ---------------------------------------------------------------------------
// Zero the workspace accumulators (counts + scalars) every call (determinism).
__global__ void vq_zero_kernel(float* __restrict__ wsF) {
  int tid = blockIdx.x * blockDim.x + threadIdx.x;
  if (tid < 2) wsF[tid] = 0.0f;
  if (tid < K_CODES) wsF[WS_COUNTS + tid] = 0.0f;
}

// Codebook prep: f32 -> f16 copy (WMMA B operand) + per-code squared norms.
__global__ void vq_prep_kernel(const float* __restrict__ cb,
                               _Float16* __restrict__ cb16,
                               float* __restrict__ c2v) {
  const int row = blockIdx.x;        // 0..1023
  const int tid = threadIdx.x;       // 256 threads, 2 elems each
  const float* src = cb + (size_t)row * D_MODEL;
  float s = 0.0f;
  #pragma unroll
  for (int j = 0; j < 2; j++) {
    int d = tid * 2 + j;
    float v = src[d];
    s += v * v;
    cb16[(size_t)row * D_MODEL + d] = (_Float16)v;
  }
  s = waveReduceSum(s);
  __shared__ float red[8];
  int lane = tid & 31, wv = tid >> 5;
  if (lane == 0) red[wv] = s;
  __syncthreads();
  if (tid == 0) {
    float t = 0.0f;
    #pragma unroll
    for (int i = 0; i < 8; i++) t += red[i];
    c2v[row] = t;
  }
}

// ---------------------------------------------------------------------------
// Fused main kernel: one block = 16 token rows. 512 threads = 16 waves.
//  phase 0: TDM DMA of x[16][512] f32 tile -> LDS stage (wave 0)
//  phase 1: row norms x2[16] from LDS
//  phase 2: WMMA GEMM: dist[16][1024] = x2 + c2 - 2*x.cT  (into LDS, f32)
//  phase 3: per-wave-per-row categorical sampling, quantized gather, stats
__global__ void __launch_bounds__(512)
vq_main_kernel(const float* __restrict__ x,
               const unsigned char* __restrict__ pad,
               const float* __restrict__ cb,
               const _Float16* __restrict__ cb16,
               const float* __restrict__ c2,
               float* __restrict__ outQ,
               float* __restrict__ outS,
               float* __restrict__ wsF) {
  extern __shared__ char smem[];
  float* dist  = (float*)smem;                              // [16][1024] f32
  float* stage = (float*)(smem + LDS_DIST_BYTES);           // [16][512]  f32
  float* x2    = (float*)(smem + LDS_DIST_BYTES + LDS_STAGE_BYTES); // [16]

  const int tid  = threadIdx.x;
  const int lane = tid & 31;
  const int wv   = tid >> 5;            // 0..15
  const int row0 = blockIdx.x * 16;

  // ---- phase 0: TDM tile DMA (issued by wave 0 only) ----------------------
  if (wv == 0) {
    const unsigned ldsOff = (unsigned)(uintptr_t)stage;  // low 32b = LDS addr
    tdm_load_f32_tile_to_lds(ldsOff, x + (size_t)row0 * D_MODEL, 16u);
    __builtin_amdgcn_s_wait_tensorcnt(0);
  }
  __syncthreads();

  // ---- phase 1: wave w computes ||x_w||^2 from the LDS stage --------------
  {
    const int r = wv;
    const float* xr = stage + r * D_MODEL;
    const int dbase = lane * 16;
    float s = 0.0f;
    #pragma unroll
    for (int j = 0; j < 16; j += 4) {
      v4f v = *(const v4f*)(xr + dbase + j);
      s += v[0] * v[0] + v[1] * v[1] + v[2] * v[2] + v[3] * v[3];
    }
    s = waveReduceSum(s);
    if (lane == 0) x2[r] = s;
  }
  __syncthreads();

  // ---- phase 2: WMMA GEMM, wave w owns columns [w*64, w*64+64) ------------
  {
    const int m       = lane & 15;        // A row / B codebook-row-in-tile
    const int koff    = (lane >> 4) * 8;  // K sub-offset per half-wave
    const int colbase = wv * 64;

    v8f acc[4];
    #pragma unroll
    for (int t = 0; t < 4; t++)
      #pragma unroll
      for (int i = 0; i < 8; i++) acc[t][i] = 0.0f;

    for (int kc = 0; kc < D_MODEL; kc += 32) {
      // A fragment (16x32 f16): convert 16 f32 from LDS stage per lane
      v16h a;
      {
        const float* pa = stage + m * D_MODEL + kc + koff;
        v4f f0 = *(const v4f*)(pa + 0);
        v4f f1 = *(const v4f*)(pa + 4);
        v4f f2 = *(const v4f*)(pa + 16);
        v4f f3 = *(const v4f*)(pa + 20);
        #pragma unroll
        for (int i = 0; i < 4; i++) {
          a[i + 0]  = (_Float16)f0[i];
          a[i + 4]  = (_Float16)f1[i];
          a[i + 8]  = (_Float16)f2[i];
          a[i + 12] = (_Float16)f3[i];
        }
      }
      #pragma unroll
      for (int t = 0; t < 4; t++) {
        // B fragment (32x16 f16): column n == codebook row (L2-resident)
        const _Float16* pb =
            cb16 + (size_t)(colbase + t * 16 + m) * D_MODEL + kc + koff;
        v8h lo = *(const v8h*)pb;
        v8h hi = *(const v8h*)(pb + 16);
        v16h b;
        #pragma unroll
        for (int i = 0; i < 8; i++) { b[i] = lo[i]; b[i + 8] = hi[i]; }
        acc[t] = __builtin_amdgcn_wmma_f32_16x16x32_f16(
            false, a, false, b, (short)0, acc[t], false, false);
      }
    }

    // D layout: VGPR v, lanes 0-15 -> M=v ; lanes 16-31 -> M=v+8
    const int rowHalf = (lane >> 4) * 8;
    #pragma unroll
    for (int t = 0; t < 4; t++) {
      const int col = colbase + t * 16 + (lane & 15);
      const float cc = c2[col];
      #pragma unroll
      for (int v = 0; v < 8; v++) {
        const int r = v + rowHalf;
        dist[r * K_CODES + col] = x2[r] + cc - 2.0f * acc[t][v];
      }
    }
  }
  __syncthreads();

  // ---- phase 3: sampling; wave w handles row w ----------------------------
  {
    const int r    = wv;
    const int grow = row0 + r;
    const bool isPad = pad[grow] != 0;
    float* drow = dist + r * K_CODES;
    float* srow = outS + (size_t)grow * K_CODES;
    float* qrow = outQ + (size_t)grow * D_MODEL;
    const float* xr = stage + r * D_MODEL;   // x row still resident in LDS
    const int cbase = lane * 32;   // 32 contiguous dist columns per lane
    const int dbase = lane * 16;   // 16 contiguous D elements per lane

    if (isPad) {
      // only code 0 allowed; codebook row 0 is zero -> quantized = 0
      #pragma unroll
      for (int j = 0; j < 32; j += 4) {
        v4f z; z[0] = z[1] = z[2] = z[3] = 0.0f;
        if (lane == 0 && j == 0) z[0] = (float)NUM_SAMPLES;
        *(v4f*)(srow + cbase + j) = z;
      }
      #pragma unroll
      for (int j = 0; j < 16; j += 4) {
        v4f z; z[0] = z[1] = z[2] = z[3] = 0.0f;
        *(v4f*)(qrow + dbase + j) = z;
      }
      if (lane == 0) {
        atomicAdd(&wsF[WS_SSE], x2[r]);                 // (0 - x)^2 summed
        atomicAdd(&wsF[WS_COUNTS + 0], (float)NUM_SAMPLES);
      }
    } else {
      // logits = -dist, code 0 disallowed
      float mx = -3.0e38f;
      #pragma unroll
      for (int j = 0; j < 32; j++) {
        const int col = cbase + j;
        if (col != 0) mx = fmaxf(mx, -drow[col]);
      }
      mx = waveReduceMax(mx);

      float ls = 0.0f;  // overwrite dist row with unnormalized probs
      #pragma unroll
      for (int j = 0; j < 32; j++) {
        const int col = cbase + j;
        const float p = (col == 0) ? 0.0f : __expf(-drow[col] - mx);
        drow[col] = p;
        ls += p;
      }
      float incl = ls;
      #pragma unroll
      for (int o = 1; o < 32; o <<= 1) {
        float tpp = __shfl_up(incl, o, 32);
        if (lane >= o) incl += tpp;
      }
      const float excl = incl - ls;
      const float Z    = __shfl(incl, 31, 32);

      // draws: lane s in [0,10) owns target t_s = u_s * Z
      float tmine = 0.0f;
      if (lane < NUM_SAMPLES) {
        unsigned u0, u1;
        threefry2x32(42u, 0u, (unsigned)grow, (unsigned)lane, u0, u1);
        const float u = ((float)(u0 >> 8) + 0.5f) * (1.0f / 16777216.0f);
        tmine = u * Z;
      }

      int idxs[NUM_SAMPLES];
      #pragma unroll
      for (int s = 0; s < NUM_SAMPLES; s++) {
        const float tv = __shfl(tmine, s, 32);
        const bool hit = (ls > 0.0f) && (tv >= excl) && (tv < excl + ls);
        const unsigned long long mk = __ballot(hit);
        const int src = mk ? (__ffsll((long long)mk) - 1) : 31;
        int idx = cbase + 31;
        if (lane == src) {
          float c = excl;
          for (int j = 0; j < 32; j++) {
            c += drow[cbase + j];
            if (tv < c) { idx = cbase + j; break; }
          }
        }
        idxs[s] = __shfl(idx, src, 32);
      }

      // samples row: every lane writes its 32 columns with counts
      #pragma unroll
      for (int j = 0; j < 32; j += 4) {
        v4f v;
        #pragma unroll
        for (int e = 0; e < 4; e++) {
          const int col = cbase + j + e;
          float cnt = 0.0f;
          #pragma unroll
          for (int s = 0; s < NUM_SAMPLES; s++) cnt += (idxs[s] == col) ? 1.0f : 0.0f;
          v[e] = cnt;
        }
        *(v4f*)(srow + cbase + j) = v;
      }
      if (lane == 0) {
        #pragma unroll
        for (int s = 0; s < NUM_SAMPLES; s++)
          atomicAdd(&wsF[WS_COUNTS + idxs[s]], 1.0f);
      }

      // quantized = mean of the 10 sampled codebook rows (f32 codebook)
      float q[16];
      #pragma unroll
      for (int j = 0; j < 16; j++) q[j] = 0.0f;
      #pragma unroll
      for (int s = 0; s < NUM_SAMPLES; s++) {
        const float* cr = cb + (size_t)idxs[s] * D_MODEL + dbase;
        #pragma unroll
        for (int j = 0; j < 16; j += 4) {
          v4f v = *(const v4f*)(cr + j);
          q[j + 0] += v[0]; q[j + 1] += v[1]; q[j + 2] += v[2]; q[j + 3] += v[3];
        }
      }
      float sse = 0.0f;
      #pragma unroll
      for (int j = 0; j < 16; j += 4) {
        v4f xv = *(const v4f*)(xr + dbase + j);
        v4f qv;
        #pragma unroll
        for (int e = 0; e < 4; e++) {
          qv[e] = q[j + e] * (1.0f / NUM_SAMPLES);
          const float d = qv[e] - xv[e];
          sse += d * d;
        }
        *(v4f*)(qrow + dbase + j) = qv;
      }
      sse = waveReduceSum(sse);
      if (lane == 0) {
        atomicAdd(&wsF[WS_SSE], sse);
        atomicAdd(&wsF[WS_NONPAD], 1.0f);
      }
    }
  }
}

// ---------------------------------------------------------------------------
__global__ void vq_finalize_kernel(const float* __restrict__ wsF,
                                   float* __restrict__ outScalars) {
  __shared__ float red[32];
  const int tid = threadIdx.x;  // 1024 threads
  const float c = wsF[WS_COUNTS + tid];
  const float p = c * (1.0f / ((float)N_ROWS * (float)NUM_SAMPLES));
  float e = p * __logf(p + 1e-10f);
  e = waveReduceSum(e);
  const int lane = tid & 31, wv = tid >> 5;
  if (lane == 0) red[wv] = e;
  __syncthreads();
  if (wv == 0) {
    float v = red[lane];
    v = waveReduceSum(v);
    if (lane == 0) {
      const float perplexity = __expf(-v);
      const float nonpad = wsF[WS_NONPAD] * (float)D_MODEL;
      const float loss = 0.25f * wsF[WS_SSE] / nonpad;
      outScalars[0] = loss;
      outScalars[1] = perplexity;
    }
  }
}

// ---------------------------------------------------------------------------
extern "C" void kernel_launch(void* const* d_in, const int* in_sizes, int n_in,
                              void* d_out, int out_size, void* d_ws, size_t ws_size,
                              hipStream_t stream) {
  (void)in_sizes; (void)n_in; (void)out_size; (void)ws_size;
  const float*         x   = (const float*)d_in[0];
  const unsigned char* pad = (const unsigned char*)d_in[1];  // bool mask (1B)
  const float*         cb  = (const float*)d_in[2];

  float*    wsF  = (float*)d_ws;
  _Float16* cb16 = (_Float16*)((char*)d_ws + WS_CB16_BYTES);

  float* outQ       = (float*)d_out;
  float* outS       = outQ + (size_t)N_ROWS * D_MODEL;
  float* outScalars = outS + (size_t)N_ROWS * K_CODES;

  vq_zero_kernel<<<4, 256, 0, stream>>>(wsF);
  vq_prep_kernel<<<K_CODES, 256, 0, stream>>>(cb, cb16, wsF + WS_C2);

  vq_main_kernel<<<N_ROWS / 16, 512, LDS_TOTAL_BYTES, stream>>>(
      x, pad, cb, cb16, wsF + WS_C2, outQ, outS, wsF);

  vq_finalize_kernel<<<1, 1024, 0, stream>>>(wsF, outScalars);
}
